// MoE_171798692232
// MI455X (gfx1250) — matrix operand order repaired and checked by reading, not compiled, not run on hardware
//
#include <hip/hip_runtime.h>
#include <hip/hip_bf16.h>

typedef __attribute__((ext_vector_type(16))) _Float16 v16h;
typedef __attribute__((ext_vector_type(8)))  float    v8f;

#define D_MODEL 1024
#define D_FF    4096
#define N_EXP   8
#define NTOK    4096   /* 4 * 1024 */

// ---------------------------------------------------------------------------
// CDNA5 async global->LDS copy (16B per lane), tracked by ASYNCcnt.
// VDST = per-lane LDS byte address (low 32 bits of generic pointer),
// VADDR = 64-bit global address, GV mode.
// ---------------------------------------------------------------------------
__device__ __forceinline__ void async_copy_b128_to_lds(const void* gptr, void* lptr) {
    unsigned long long g = (unsigned long long)(uintptr_t)gptr;
    unsigned int       l = (unsigned int)(uintptr_t)lptr;
    asm volatile("global_load_async_to_lds_b128 %0, %1, off" :: "v"(l), "v"(g) : "memory");
}
__device__ __forceinline__ void wait_asynccnt0() {
    asm volatile("s_wait_asynccnt 0x0" ::: "memory");
}

// ---------------------------------------------------------------------------
// Small utility kernels
// ---------------------------------------------------------------------------
__global__ void zero_f32_kernel(float* __restrict__ p, int n) {
    int i = blockIdx.x * blockDim.x + threadIdx.x;
    int stride = gridDim.x * blockDim.x;
    for (; i < n; i += stride) p[i] = 0.0f;
}

__global__ void cast_f32_to_f16_kernel(const float* __restrict__ src,
                                       _Float16* __restrict__ dst, int n) {
    int i = blockIdx.x * blockDim.x + threadIdx.x;
    int stride = gridDim.x * blockDim.x;
    for (; i < n; i += stride) dst[i] = (_Float16)src[i];
}

// ---------------------------------------------------------------------------
// Gating: logits = x @ Wg + bg ; softmax ; top-2 mask ; renormalize.
// Exact f32 (tiny cost: ~67 MFLOP). One thread per token.
// ---------------------------------------------------------------------------
__global__ __launch_bounds__(256)
void gate_kernel(const float* __restrict__ x, const float* __restrict__ Wg,
                 const float* __restrict__ bg, float* __restrict__ probs) {
    int t = blockIdx.x * blockDim.x + threadIdx.x;
    if (t >= NTOK) return;
    float acc[N_EXP];
#pragma unroll
    for (int e = 0; e < N_EXP; ++e) acc[e] = bg[e];
    const float* xr = x + (size_t)t * D_MODEL;
    for (int d = 0; d < D_MODEL; ++d) {
        float xv = xr[d];
        const float* wr = Wg + (size_t)d * N_EXP;
#pragma unroll
        for (int e = 0; e < N_EXP; ++e) acc[e] = fmaf(xv, wr[e], acc[e]);
    }
    float mx = acc[0];
#pragma unroll
    for (int e = 1; e < N_EXP; ++e) mx = fmaxf(mx, acc[e]);
    float p[N_EXP];
#pragma unroll
    for (int e = 0; e < N_EXP; ++e) p[e] = __expf(acc[e] - mx);
    int i1 = 0;
#pragma unroll
    for (int e = 1; e < N_EXP; ++e) if (p[e] > p[i1]) i1 = e;
    int i2 = (i1 == 0) ? 1 : 0;
#pragma unroll
    for (int e = 0; e < N_EXP; ++e) if (e != i1 && p[e] > p[i2]) i2 = e;
    float inv = 1.0f / (p[i1] + p[i2]);
#pragma unroll
    for (int e = 0; e < N_EXP; ++e)
        probs[(size_t)t * N_EXP + e] = (e == i1 || e == i2) ? p[e] * inv : 0.0f;
}

// ---------------------------------------------------------------------------
// Tiled WMMA GEMM:  C[M,N] = A_f16[M,K] * B_f32[K,N]  (B converted on the fly)
//   EPI==0: Hout = f16( gelu(C + bias[n]) )
//   EPI==1: Fout[m,n] += probs[m][expert] * (C + bias[n])
// Block tile 128x128x32, 8 waves, each wave 32x64 (2x4 wmma tiles).
// Double-buffered LDS; A tile via async global->LDS (no VGPR staging),
// B tile via register staging with f32->f16 transpose.
// ---------------------------------------------------------------------------
#define BM 128
#define BN 128
#define BK 32
#define LDS_STRIDE 40   // halves per row (32 + 8 pad)

__device__ __forceinline__ v16h load_frag16(const _Float16* base, int off0, int off1) {
    union { v16h v; uint4 q[2]; } f;
    f.q[0] = *(const uint4*)(base + off0);
    f.q[1] = *(const uint4*)(base + off1);
    return f.v;
}

template <int EPI>
__global__ __launch_bounds__(256)
void moe_gemm_kernel(const _Float16* __restrict__ A,   // [M,K] f16 row-major
                     const float*   __restrict__ Bw,   // [K,N] f32 row-major
                     const float*   __restrict__ bias, // [N]
                     _Float16*      __restrict__ Hout, // EPI==0 output [M,N]
                     float*         __restrict__ Fout, // EPI==1 output [M,N]
                     const float*   __restrict__ probs,// [M,8]
                     int expert, int M, int N, int K) {
    __shared__ _Float16 As[2][BM * LDS_STRIDE] __attribute__((aligned(16)));
    __shared__ _Float16 Bs[2][BN * LDS_STRIDE] __attribute__((aligned(16)));
    __shared__ float    Pr[BM];

    const int tid   = threadIdx.x;
    const int lane  = tid & 31;
    const int wid   = tid >> 5;
    const int waveM = wid & 3;   // 4 strips of 32 rows
    const int waveN = wid >> 2;  // 2 strips of 64 cols
    const int lrow  = lane & 15;
    const int half  = lane >> 4;

    const int mBase = blockIdx.y * BM;
    const int nBase = blockIdx.x * BN;

    if (EPI == 1) {
        if (tid < BM) Pr[tid] = probs[(size_t)(mBase + tid) * N_EXP + expert];
    }

    // ---- per-thread staging indices (fixed across k-steps) ----
    const _Float16* aSrc[2];  int aDst[2];
#pragma unroll
    for (int i = 0; i < 2; ++i) {
        int lin = tid + i * 256;            // 0..511
        int r = lin >> 2;                   // 0..127 row
        int ck = lin & 3;                   // chunk of 8 halves
        aSrc[i] = A + (size_t)(mBase + r) * K + ck * 8;
        aDst[i] = r * LDS_STRIDE + ck * 8;
    }
    const float* bSrc[4];  int bDst[4];
#pragma unroll
    for (int i = 0; i < 4; ++i) {
        int lin = tid + i * 256;            // 0..1023
        int kr = lin >> 5;                  // 0..31  K row
        int nb = (lin & 31) * 4;            // N col group of 4
        bSrc[i] = Bw + (size_t)kr * N + nBase + nb;
        bDst[i] = nb * LDS_STRIDE + kr;     // Bs[n][k] layout
    }

    float4 bReg[4];

    // A tile: async global->LDS, no VGPR staging
    auto prefetchA = [&](int k0, int buf) {
#pragma unroll
        for (int i = 0; i < 2; ++i)
            async_copy_b128_to_lds(aSrc[i] + k0, &As[buf][aDst[i]]);
    };
    // B tile: stage f32 in regs (overlap), convert+transpose later
    auto prefetchB = [&](int k0) {
        size_t rowOff = (size_t)k0 * N;
#pragma unroll
        for (int i = 0; i < 4; ++i) bReg[i] = *(const float4*)(bSrc[i] + rowOff);
    };
    auto stageB = [&](int buf) {
#pragma unroll
        for (int i = 0; i < 4; ++i) {
            float4 v = bReg[i];
            _Float16* p = &Bs[buf][bDst[i]];
            p[0 * LDS_STRIDE] = (_Float16)v.x;
            p[1 * LDS_STRIDE] = (_Float16)v.y;
            p[2 * LDS_STRIDE] = (_Float16)v.z;
            p[3 * LDS_STRIDE] = (_Float16)v.w;
        }
    };

    v8f c[2][4];
#pragma unroll
    for (int tm = 0; tm < 2; ++tm)
#pragma unroll
        for (int tn = 0; tn < 4; ++tn) c[tm][tn] = (v8f){};

    const int nSteps = K / BK;
    prefetchA(0, 0);
    prefetchB(0);
    stageB(0);
    wait_asynccnt0();
    __syncthreads();

    for (int s = 0; s < nSteps; ++s) {
        const int cur = s & 1;
        if (s + 1 < nSteps) {
            prefetchA((s + 1) * BK, 1 - cur);  // async, no regs; overlaps compute
            prefetchB((s + 1) * BK);           // issue loads; consumed in stageB
        }

        // ---- fragments from buf[cur] (ISA 7.12.2 16-bit layouts) ----
        v16h af[2], bf[4];
#pragma unroll
        for (int tm = 0; tm < 2; ++tm) {
            int row = waveM * 32 + tm * 16 + lrow;
            const _Float16* base = &As[cur][row * LDS_STRIDE];
            af[tm] = load_frag16(base, 8 * half, 16 + 8 * half);
        }
#pragma unroll
        for (int tn = 0; tn < 4; ++tn) {
            int col = waveN * 64 + tn * 16 + lrow;
            const _Float16* base = &Bs[cur][col * LDS_STRIDE];
            bf[tn] = load_frag16(base, 8 * half, 16 + 8 * half);
        }
#pragma unroll
        for (int tm = 0; tm < 2; ++tm)
#pragma unroll
            for (int tn = 0; tn < 4; ++tn)
                c[tm][tn] = __builtin_amdgcn_wmma_f32_16x16x32_f16(
                    false, af[tm], false, bf[tn], (short)0, c[tm][tn], false, false);

        if (s + 1 < nSteps) stageB(1 - cur);  // prev barrier retired reads of buf[1-cur]
        wait_asynccnt0();                     // A tile for next step landed in LDS
        __syncthreads();
    }

    // ---- epilogue ----
#pragma unroll
    for (int tm = 0; tm < 2; ++tm) {
#pragma unroll
        for (int tn = 0; tn < 4; ++tn) {
            int col = nBase + waveN * 64 + tn * 16 + lrow;
            float bv = bias[col];
#pragma unroll
            for (int r = 0; r < 8; ++r) {
                int rowLoc = waveM * 32 + tm * 16 + half * 8 + r;
                int row = mBase + rowLoc;
                float v = c[tm][tn][r] + bv;
                if (EPI == 0) {
                    float g = 0.5f * v * (1.0f + erff(v * 0.70710678118654752f));
                    Hout[(size_t)row * N + col] = (_Float16)g;
                } else {
                    Fout[(size_t)row * N + col] += Pr[rowLoc] * v;
                }
            }
        }
    }
}

// ---------------------------------------------------------------------------
// Launcher
// ---------------------------------------------------------------------------
extern "C" void kernel_launch(void* const* d_in, const int* in_sizes, int n_in,
                              void* d_out, int out_size, void* d_ws, size_t ws_size,
                              hipStream_t stream) {
    const float* x  = (const float*)d_in[0];
    const float* W1 = (const float*)d_in[1];
    const float* b1 = (const float*)d_in[2];
    const float* W2 = (const float*)d_in[3];
    const float* b2 = (const float*)d_in[4];
    const float* Wg = (const float*)d_in[5];
    const float* bg = (const float*)d_in[6];
    float* out = (float*)d_out;

    // Workspace layout
    char* ws = (char*)d_ws;
    float*    probs = (float*)ws;                                  // 4096*8*4   = 128 KB
    _Float16* xh    = (_Float16*)(ws + 131072);                    // 4096*1024*2 = 8 MB
    _Float16* Hh    = (_Float16*)(ws + 131072 + 8388608);          // 4096*4096*2 = 33.5 MB

    const int nOut = NTOK * D_MODEL;            // 4,194,304
    zero_f32_kernel<<<1024, 256, 0, stream>>>(out, nOut);
    cast_f32_to_f16_kernel<<<2048, 256, 0, stream>>>(x, xh, NTOK * D_MODEL);
    gate_kernel<<<NTOK / 256, 256, 0, stream>>>(x, Wg, bg, probs);

    for (int e = 0; e < N_EXP; ++e) {
        const float* W1e = W1 + (size_t)e * D_MODEL * D_FF;
        const float* b1e = b1 + (size_t)e * D_FF;
        const float* W2e = W2 + (size_t)e * D_FF * D_MODEL;
        const float* b2e = b2 + (size_t)e * D_MODEL;

        // H = f16(gelu(xh @ W1e + b1e))   [4096 x 4096]
        moe_gemm_kernel<0><<<dim3(D_FF / BN, NTOK / BM), 256, 0, stream>>>(
            xh, W1e, b1e, Hh, nullptr, nullptr, 0, NTOK, D_FF, D_MODEL);

        // out += probs[:,e] * (H @ W2e + b2e)   [4096 x 1024]
        moe_gemm_kernel<1><<<dim3(D_MODEL / BN, NTOK / BM), 256, 0, stream>>>(
            Hh, W2e, b2e, nullptr, out, probs, e, NTOK, D_MODEL, D_FF);
    }
}
